// BidirRecurrentModel_64476049047577
// MI455X (gfx1250) — matrix operand order repaired and checked by reading, not compiled, not run on hardware
//
#include <hip/hip_runtime.h>
#include <hip/hip_bf16.h>

// ---------------------------------------------------------------------------
// Problem constants (from the reference)
// ---------------------------------------------------------------------------
#define TT    2048   // T
#define BB    64     // batch
#define HH    256    // hidden
#define DIN   128
#define DOUT  128
#define G4H   1024   // 4*H

#define NWG      16    // persistent workgroups (each owns H/16 = 16 hidden units)
#define NTHREADS 256   // 8 wave32s

typedef __attribute__((ext_vector_type(16))) __bf16 v16bf;
typedef __attribute__((ext_vector_type(8)))  __bf16 v8bf;
typedef __attribute__((ext_vector_type(8)))  float  v8f;
typedef __attribute__((ext_vector_type(4)))  int    v4i;

// CDNA5 async global->LDS path (types verified against hipcc diagnostics).
#if __has_builtin(__builtin_amdgcn_global_load_async_to_lds_b128) && \
    __has_builtin(__builtin_amdgcn_s_wait_asynccnt)
#define USE_ASYNC_LDS 1
typedef __attribute__((address_space(1))) v4i* gv4i_p;
typedef __attribute__((address_space(3))) v4i* lv4i_p;
#endif

// ---------------------------------------------------------------------------
// WMMA helpers (wave32; layouts per CDNA5 ISA 7.12.2)
// ---------------------------------------------------------------------------
__device__ __forceinline__ v8f wmma_bf16(v16bf a, v16bf b, v8f c) {
  return __builtin_amdgcn_wmma_f32_16x16x32_bf16(false, a, false, b,
                                                 (short)0, c, false, false);
}

union Frag16 { v16bf v; v8bf h[2]; };

// A fragment: 16x32 bf16, row-major source, leading dim `ld` (halves).
__device__ __forceinline__ v16bf load_a_frag(const __bf16* base, int ld,
                                             int mt, int kt, int lane) {
  int m  = mt + (lane & 15);
  int k0 = kt + ((lane >> 4) << 3);
  const __bf16* p = base + (size_t)m * ld + k0;
  Frag16 f;
  f.h[0] = *(const v8bf*)(p);
  f.h[1] = *(const v8bf*)(p + 16);
  return f.v;
}

// B fragment: 32x16 (KxN) from row-major W(N,K): B[k][n] = W[n][k].
__device__ __forceinline__ v16bf load_b_frag(const __bf16* w, int ld,
                                             int nt, int kt, int lane) {
  int n  = nt + (lane & 15);
  int k0 = kt + ((lane >> 4) << 3);
  const __bf16* p = w + (size_t)n * ld + k0;
  Frag16 f;
  f.h[0] = *(const v8bf*)(p);
  f.h[1] = *(const v8bf*)(p + 16);
  return f.v;
}

// ---------------------------------------------------------------------------
// Staging: global bf16 -> LDS via CDNA5 async DMA (ASYNCcnt tracked)
// ---------------------------------------------------------------------------
__device__ __forceinline__ void stage_bf16(__bf16* dst, const __bf16* src,
                                           int halves) {
  int n16 = halves >> 3;  // 16-byte chunks
#if defined(USE_ASYNC_LDS)
  for (int i = threadIdx.x; i < n16; i += NTHREADS) {
    __builtin_amdgcn_global_load_async_to_lds_b128(
        (gv4i_p)((const v4i*)src + i), (lv4i_p)((v4i*)dst + i), 0, 0);
  }
  __builtin_amdgcn_s_wait_asynccnt(0);
#else
  const uint4* s4 = (const uint4*)src;
  uint4*       d4 = (uint4*)dst;
  for (int i = threadIdx.x; i < n16; i += NTHREADS) d4[i] = s4[i];
#endif
}

__device__ __forceinline__ float sigm(float x) {
  return 1.0f / (1.0f + __expf(-x));
}

// Counter-based grid barrier across the NWG persistent workgroups.
__device__ __forceinline__ void grid_sync(unsigned* bar, unsigned* epoch) {
  __syncthreads();
  if (threadIdx.x == 0) {
    __threadfence();
    atomicAdd(bar, 1u);
    *epoch += NWG;
    volatile unsigned* vb = (volatile unsigned*)bar;
    while (*vb < *epoch) __builtin_amdgcn_s_sleep(2);
    __threadfence();
  }
  __syncthreads();
}

// ---------------------------------------------------------------------------
// Prep kernels
// ---------------------------------------------------------------------------
__global__ void k_convert_bf16(__bf16* dst, const float* src, int n) {
  int i = blockIdx.x * blockDim.x + threadIdx.x;
  if (i < n) dst[i] = (__bf16)src[i];
}

__global__ void k_bias_combine(float* b0, const float* bx0, const float* bh0,
                               float* b1, const float* bx1, const float* bh1) {
  int i = blockIdx.x * blockDim.x + threadIdx.x;
  if (i < G4H) {
    b0[i] = bx0[i] + bh0[i];
    b1[i] = bx1[i] + bh1[i];
  }
}

__global__ void k_zero_state(unsigned* bar, __bf16* hz, int halves) {
  int i = blockIdx.x * blockDim.x + threadIdx.x;
  if (i == 0) bar[0] = 0u;
  if (i < halves) hz[i] = (__bf16)0.0f;
}

// ---------------------------------------------------------------------------
// Persistent fused 2-layer LSTM scan + backward one-step + FC.
// Workgroup wg owns hidden units [wg*16, wg*16+16) of both layers (64 gate
// columns).  Weight slices stay resident in LDS for the whole kernel.
// Per wave: one N-tile, two M-tiles (mi, mi+2) with dual accumulators so each
// B fragment feeds two independent back-to-back WMMAs (hides LDS latency).
// ---------------------------------------------------------------------------
__global__ void lstm_persistent(
    const float* __restrict__ x,                       // (B, T, DIN) f32
    const __bf16* __restrict__ wx0, const __bf16* __restrict__ wh0,
    const __bf16* __restrict__ wx1, const __bf16* __restrict__ wh1,
    const __bf16* __restrict__ fcw,                    // (DOUT, 2H) bf16
    const float* __restrict__ fcb,                     // (DOUT) f32
    const float* __restrict__ b0c, const float* __restrict__ b1c,  // (4H)
    __bf16* __restrict__ h0buf,                        // 2 * (B*H), parity t&1
    __bf16* __restrict__ h1buf,                        // 2 * (B*H)
    __bf16* __restrict__ hb0, __bf16* __restrict__ hb1,// (B*H) each
    float* __restrict__ out,                           // (B, DOUT)
    unsigned* __restrict__ bar) {
  extern __shared__ char smem[];
  __bf16* sWx0 = (__bf16*)smem;            // 64 x 128
  __bf16* sWh0 = sWx0 + 64 * DIN;          // 64 x 256
  __bf16* sWx1 = sWh0 + 64 * HH;           // 64 x 256
  __bf16* sWh1 = sWx1 + 64 * HH;           // 64 x 256
  __bf16* sX   = sWh1 + 64 * HH;           // 64 x 128 (x_t, bf16)
  __bf16* sH0  = sX   + BB * DIN;          // 64 x 256
  __bf16* sH1  = sH0  + BB * HH;           // 64 x 256
  float*  sPre = (float*)(sH1 + BB * HH);  // 64 x 64 pre-activations (f32)
  float*  sC0  = sPre + 64 * 64;           // 64 x 16 cell state, layer 0
  float*  sC1  = sC0  + BB * 16;           // 64 x 16 cell state, layer 1
  float*  sB0  = sC1  + BB * 16;           // 64 combined biases layer 0
  float*  sB1  = sB0  + 64;                // 64 combined biases layer 1

  const int wg      = blockIdx.x;
  const int lane    = threadIdx.x & 31;
  const int wave    = threadIdx.x >> 5;
  const int colbase = wg * 16;

  // Per-wave tile assignment: N-tile nj, M-tiles mi0 and mi0+2.
  const int nt  = (wave & 3) * 16;
  const int mt0 = (wave >> 2) * 16;        // 0 or 16
  const int mt1 = mt0 + 32;                // 32 or 48
  const int nlc = nt + (lane & 15);        // this lane's local output column
  const int mb0 = mt0 + ((lane >> 4) << 3);
  const int mb1 = mt1 + ((lane >> 4) << 3);

  // ---- one-time: pull this WG's weight slices into LDS -------------------
  {
    auto load_slice = [&](__bf16* dst, const __bf16* src, int K) {
      int kc = K >> 3;
      int chunks = 64 * kc;
      for (int i = threadIdx.x; i < chunks; i += NTHREADS) {
        int j = i / kc, c = i % kc;
        int grow = ((j >> 4) * HH) + colbase + (j & 15);
        ((v8bf*)dst)[i] = ((const v8bf*)(src + (size_t)grow * K))[c];
      }
    };
    load_slice(sWx0, wx0, DIN);
    load_slice(sWh0, wh0, HH);
    load_slice(sWx1, wx1, HH);
    load_slice(sWh1, wh1, HH);
    if (threadIdx.x < 64) {
      int j = threadIdx.x;
      int grow = ((j >> 4) * HH) + colbase + (j & 15);
      sB0[j] = b0c[grow];
      sB1[j] = b1c[grow];
    }
    for (int i = threadIdx.x; i < 2 * BB * 16; i += NTHREADS) sC0[i] = 0.0f;
  }
  __syncthreads();

  unsigned epoch = 0;

  // ======================= the T-step recurrence =========================
  for (int t = 0; t < TT; ++t) {
    const int par = t & 1, parp = par ^ 1;

    // -------- Phase A: layer-0:  pre = x_t Wx0^T + h0 Wh0^T + b0 ---------
    // Stage x_t (f32 -> bf16), vectorized float4 -> b64 LDS stores.
    for (int i = threadIdx.x; i < (BB * DIN) / 4; i += NTHREADS) {
      int b = i >> 5, d4 = i & 31;                     // 32 float4 per row
      size_t gi = ((size_t)b * TT + t) * DIN + d4 * 4;
      float4 v = *(const float4*)&x[gi];
      union { __bf16 h[4]; uint2 u; } o;
      o.h[0] = (__bf16)v.x; o.h[1] = (__bf16)v.y;
      o.h[2] = (__bf16)v.z; o.h[3] = (__bf16)v.w;
      *(uint2*)&sX[i * 4] = o.u;
      if (t + 1 < TT) __builtin_prefetch(&x[gi + DIN], 0, 1);  // next step
    }
    stage_bf16(sH0, h0buf + (size_t)parp * (BB * HH), BB * HH);
    __syncthreads();

    {
      v8f acc0 = {}, acc1 = {};
#pragma unroll
      for (int kb = 0; kb < 4; ++kb) {
        v16bf b  = load_b_frag(sWx0, DIN, nt, kb * 32, lane);
        v16bf a0 = load_a_frag(sX, DIN, mt0, kb * 32, lane);
        v16bf a1 = load_a_frag(sX, DIN, mt1, kb * 32, lane);
        acc0 = wmma_bf16(a0, b, acc0);
        acc1 = wmma_bf16(a1, b, acc1);
      }
#pragma unroll
      for (int kb = 0; kb < 8; ++kb) {
        v16bf b  = load_b_frag(sWh0, HH, nt, kb * 32, lane);
        v16bf a0 = load_a_frag(sH0, HH, mt0, kb * 32, lane);
        v16bf a1 = load_a_frag(sH0, HH, mt1, kb * 32, lane);
        acc0 = wmma_bf16(a0, b, acc0);
        acc1 = wmma_bf16(a1, b, acc1);
      }
      const float bias = sB0[nlc];
#pragma unroll
      for (int r = 0; r < 8; ++r) {
        sPre[(mb0 + r) * 64 + nlc] = acc0[r] + bias;
        sPre[(mb1 + r) * 64 + nlc] = acc1[r] + bias;
      }
    }
    __syncthreads();

    {
      __bf16* h0w = h0buf + (size_t)par * (BB * HH);
#pragma unroll
      for (int r = 0; r < 4; ++r) {
        int e = threadIdx.x + NTHREADS * r;            // 0..1023 = (b,u)
        int b = e >> 4, u = e & 15;
        const float* pr = sPre + b * 64;
        float f  = pr[u], ii = pr[16 + u], ct = pr[32 + u], o = pr[48 + u];
        float c  = sigm(f) * sC0[b * 16 + u] + sigm(ii) * tanhf(ct);
        sC0[b * 16 + u] = c;
        h0w[b * HH + colbase + u] = (__bf16)(sigm(o) * tanhf(c));
      }
    }

    grid_sync(bar, &epoch);   // all h0(t) slices visible everywhere

    // -------- Phase B: layer-1:  pre = h0(t) Wx1^T + h1 Wh1^T + b1 -------
    stage_bf16(sH0, h0buf + (size_t)par * (BB * HH), BB * HH);
    stage_bf16(sH1, h1buf + (size_t)parp * (BB * HH), BB * HH);
    __syncthreads();

    {
      v8f acc0 = {}, acc1 = {};
#pragma unroll
      for (int kb = 0; kb < 8; ++kb) {
        v16bf b  = load_b_frag(sWx1, HH, nt, kb * 32, lane);
        v16bf a0 = load_a_frag(sH0, HH, mt0, kb * 32, lane);
        v16bf a1 = load_a_frag(sH0, HH, mt1, kb * 32, lane);
        acc0 = wmma_bf16(a0, b, acc0);
        acc1 = wmma_bf16(a1, b, acc1);
      }
#pragma unroll
      for (int kb = 0; kb < 8; ++kb) {
        v16bf b  = load_b_frag(sWh1, HH, nt, kb * 32, lane);
        v16bf a0 = load_a_frag(sH1, HH, mt0, kb * 32, lane);
        v16bf a1 = load_a_frag(sH1, HH, mt1, kb * 32, lane);
        acc0 = wmma_bf16(a0, b, acc0);
        acc1 = wmma_bf16(a1, b, acc1);
      }
      const float bias = sB1[nlc];
#pragma unroll
      for (int r = 0; r < 8; ++r) {
        sPre[(mb0 + r) * 64 + nlc] = acc0[r] + bias;
        sPre[(mb1 + r) * 64 + nlc] = acc1[r] + bias;
      }
    }
    __syncthreads();

    {
      __bf16* h1w = h1buf + (size_t)par * (BB * HH);
#pragma unroll
      for (int r = 0; r < 4; ++r) {
        int e = threadIdx.x + NTHREADS * r;
        int b = e >> 4, u = e & 15;
        const float* pr = sPre + b * 64;
        float f  = pr[u], ii = pr[16 + u], ct = pr[32 + u], o = pr[48 + u];
        float c  = sigm(f) * sC1[b * 16 + u] + sigm(ii) * tanhf(ct);
        sC1[b * 16 + u] = c;
        h1w[b * HH + colbase + u] = (__bf16)(sigm(o) * tanhf(c));
      }
    }
    // No trailing barrier: next Phase A's barrier orders everything.
  }

  // =============== backward direction: ONE step from zero state ==========
  // hb0 = gates(x[T-1] Wx0^T + b0), c=0
  for (int i = threadIdx.x; i < (BB * DIN) / 4; i += NTHREADS) {
    int b = i >> 5, d4 = i & 31;
    size_t gi = ((size_t)b * TT + (TT - 1)) * DIN + d4 * 4;
    float4 v = *(const float4*)&x[gi];
    union { __bf16 h[4]; uint2 u; } o;
    o.h[0] = (__bf16)v.x; o.h[1] = (__bf16)v.y;
    o.h[2] = (__bf16)v.z; o.h[3] = (__bf16)v.w;
    *(uint2*)&sX[i * 4] = o.u;
  }
  __syncthreads();
  {
    v8f acc0 = {}, acc1 = {};
#pragma unroll
    for (int kb = 0; kb < 4; ++kb) {
      v16bf b  = load_b_frag(sWx0, DIN, nt, kb * 32, lane);
      v16bf a0 = load_a_frag(sX, DIN, mt0, kb * 32, lane);
      v16bf a1 = load_a_frag(sX, DIN, mt1, kb * 32, lane);
      acc0 = wmma_bf16(a0, b, acc0);
      acc1 = wmma_bf16(a1, b, acc1);
    }
    const float bias = sB0[nlc];
#pragma unroll
    for (int r = 0; r < 8; ++r) {
      sPre[(mb0 + r) * 64 + nlc] = acc0[r] + bias;
      sPre[(mb1 + r) * 64 + nlc] = acc1[r] + bias;
    }
  }
  __syncthreads();
#pragma unroll
  for (int r = 0; r < 4; ++r) {
    int e = threadIdx.x + NTHREADS * r;
    int b = e >> 4, u = e & 15;
    const float* pr = sPre + b * 64;
    float c = sigm(pr[16 + u]) * tanhf(pr[32 + u]);      // c_old = 0
    hb0[b * HH + colbase + u] = (__bf16)(sigm(pr[48 + u]) * tanhf(c));
  }
  grid_sync(bar, &epoch);

  // hb1 = gates(hb0 Wx1^T + b1), c=0
  stage_bf16(sH0, hb0, BB * HH);
  __syncthreads();
  {
    v8f acc0 = {}, acc1 = {};
#pragma unroll
    for (int kb = 0; kb < 8; ++kb) {
      v16bf b  = load_b_frag(sWx1, HH, nt, kb * 32, lane);
      v16bf a0 = load_a_frag(sH0, HH, mt0, kb * 32, lane);
      v16bf a1 = load_a_frag(sH0, HH, mt1, kb * 32, lane);
      acc0 = wmma_bf16(a0, b, acc0);
      acc1 = wmma_bf16(a1, b, acc1);
    }
    const float bias = sB1[nlc];
#pragma unroll
    for (int r = 0; r < 8; ++r) {
      sPre[(mb0 + r) * 64 + nlc] = acc0[r] + bias;
      sPre[(mb1 + r) * 64 + nlc] = acc1[r] + bias;
    }
  }
  __syncthreads();
#pragma unroll
  for (int r = 0; r < 4; ++r) {
    int e = threadIdx.x + NTHREADS * r;
    int b = e >> 4, u = e & 15;
    const float* pr = sPre + b * 64;
    float c = sigm(pr[16 + u]) * tanhf(pr[32 + u]);
    hb1[b * HH + colbase + u] = (__bf16)(sigm(pr[48 + u]) * tanhf(c));
  }
  grid_sync(bar, &epoch);

  // ================ FC: out = [hT | hb] @ fc_w^T + fc_b ==================
  stage_bf16(sH0, h1buf + (size_t)((TT - 1) & 1) * (BB * HH), BB * HH); // hT
  stage_bf16(sH1, hb1, BB * HH);
  __syncthreads();

  // 32 output tiles (4 M x 8 N) over 16 WGs; waves 0 and 4 take one each.
  if (wave == 0 || wave == 4) {
    const int tt = wg * 2 + (wave == 4 ? 1 : 0);
    const int mt = (tt >> 3) * 16;
    const int nc = (tt & 7) * 16;
    v8f acc = {};
#pragma unroll
    for (int kb = 0; kb < 8; ++kb)            // hT part (K = 0..255)
      acc = wmma_bf16(load_a_frag(sH0, HH, mt, kb * 32, lane),
                      load_b_frag(fcw, 2 * HH, nc, kb * 32, lane), acc);
#pragma unroll
    for (int kb = 0; kb < 8; ++kb)            // hb part (K = 256..511)
      acc = wmma_bf16(load_a_frag(sH1, HH, mt, kb * 32, lane),
                      load_b_frag(fcw, 2 * HH, nc, 256 + kb * 32, lane), acc);
    const int n  = nc + (lane & 15);
    const int mb = mt + ((lane >> 4) << 3);
    const float bias = fcb[n];
#pragma unroll
    for (int r = 0; r < 8; ++r) out[(mb + r) * DOUT + n] = acc[r] + bias;
  }
}

// ---------------------------------------------------------------------------
// Launcher
// ---------------------------------------------------------------------------
extern "C" void kernel_launch(void* const* d_in, const int* in_sizes, int n_in,
                              void* d_out, int out_size, void* d_ws,
                              size_t ws_size, hipStream_t stream) {
  (void)in_sizes; (void)n_in; (void)out_size; (void)ws_size;

  const float* x   = (const float*)d_in[0];
  const float* Wx0 = (const float*)d_in[1];
  const float* bx0 = (const float*)d_in[2];
  const float* Wh0 = (const float*)d_in[3];
  const float* bh0 = (const float*)d_in[4];
  const float* Wx1 = (const float*)d_in[5];
  const float* bx1 = (const float*)d_in[6];
  const float* Wh1 = (const float*)d_in[7];
  const float* bh1 = (const float*)d_in[8];
  const float* fcw = (const float*)d_in[9];
  const float* fcb = (const float*)d_in[10];

  char*  ws  = (char*)d_ws;
  size_t off = 0;
  auto carve = [&](size_t bytes) -> void* {
    void* p = ws + off;
    off = (off + bytes + 255) & ~(size_t)255;
    return p;
  };

  unsigned* bar = (unsigned*)carve(256);
  __bf16* hz    = (__bf16*)carve((size_t)6 * BB * HH * sizeof(__bf16));
  __bf16* h0buf = hz;                 // 2 * B*H (parity buffers)
  __bf16* h1buf = hz + 2 * BB * HH;   // 2 * B*H
  __bf16* hb0   = hz + 4 * BB * HH;   // B*H
  __bf16* hb1   = hz + 5 * BB * HH;   // B*H
  __bf16* wx0b  = (__bf16*)carve((size_t)G4H * DIN * sizeof(__bf16));
  __bf16* wh0b  = (__bf16*)carve((size_t)G4H * HH * sizeof(__bf16));
  __bf16* wx1b  = (__bf16*)carve((size_t)G4H * HH * sizeof(__bf16));
  __bf16* wh1b  = (__bf16*)carve((size_t)G4H * HH * sizeof(__bf16));
  __bf16* fcwb  = (__bf16*)carve((size_t)DOUT * 2 * HH * sizeof(__bf16));
  float*  b0c   = (float*)carve(G4H * sizeof(float));
  float*  b1c   = (float*)carve(G4H * sizeof(float));

  const int hzn = 6 * BB * HH;
  k_zero_state<<<(hzn + 255) / 256, 256, 0, stream>>>(bar, hz, hzn);
  k_convert_bf16<<<(G4H * DIN + 255) / 256, 256, 0, stream>>>(wx0b, Wx0, G4H * DIN);
  k_convert_bf16<<<(G4H * HH + 255) / 256, 256, 0, stream>>>(wh0b, Wh0, G4H * HH);
  k_convert_bf16<<<(G4H * HH + 255) / 256, 256, 0, stream>>>(wx1b, Wx1, G4H * HH);
  k_convert_bf16<<<(G4H * HH + 255) / 256, 256, 0, stream>>>(wh1b, Wh1, G4H * HH);
  k_convert_bf16<<<(DOUT * 2 * HH + 255) / 256, 256, 0, stream>>>(fcwb, fcw, DOUT * 2 * HH);
  k_bias_combine<<<(G4H + 255) / 256, 256, 0, stream>>>(b0c, bx0, bh0, b1c, bx1, bh1);

  // LDS: 4 weight slices + x_t + h0 + h1 (bf16) + pre/c/bias (f32) ~= 216.5 KB
  const size_t smem_bytes =
      (size_t)(64 * DIN + 3 * 64 * HH + BB * DIN + 2 * BB * HH) * sizeof(__bf16) +
      (size_t)(64 * 64 + 2 * BB * 16 + 128) * sizeof(float);

  lstm_persistent<<<NWG, NTHREADS, smem_bytes, stream>>>(
      x, wx0b, wh0b, wx1b, wh1b, fcwb, fcb, b0c, b1c,
      h0buf, h1buf, hb0, hb1, (float*)d_out, bar);
}
